// L0Perception_29540785062019
// MI455X (gfx1250) — matrix operand order repaired and checked by reading, not compiled, not run on hardware
//
#include <hip/hip_runtime.h>
#include <hip/hip_bf16.h>

typedef __attribute__((ext_vector_type(16))) _Float16 v16h;
typedef __attribute__((ext_vector_type(8)))  _Float16 v8h;
typedef __attribute__((ext_vector_type(8)))  float    v8f;

// ---------------------------------------------------------------------------
// WMMA fragment helpers (16x16x32 f16, wave32)
// A operand: lane = row (M = lane&15); elems 0..7 = K kb..kb+7, 8..15 = K kb+16..kb+23, kb = (lane>=16)?8:0
// B operand: lane = col (N = lane&15); elems 0..7 = K kb..kb+7, 8..15 = K kb+16..kb+23, kb = (lane>=16)?8:0
// C/D:       lane = col; elem r = row ((lane>=16)?8:0)+r
// ---------------------------------------------------------------------------
__device__ __forceinline__ v16h load_fragA(const _Float16* rowptr, int lane) {
  const int kb = (lane >> 4) << 3;
  v8h lo = *(const v8h*)(rowptr + kb);
  v8h hi = *(const v8h*)(rowptr + kb + 16);
  v16h r;
#pragma unroll
  for (int i = 0; i < 8; ++i) { r[i] = lo[i]; r[i + 8] = hi[i]; }
  return r;
}

__device__ __forceinline__ v16h load_fragB(const _Float16* colptr, int lane) {
  const int kb = (lane >> 4) << 3;
  v8h lo = *(const v8h*)(colptr + kb);
  v8h hi = *(const v8h*)(colptr + kb + 16);
  v16h r;
#pragma unroll
  for (int i = 0; i < 8; ++i) { r[i] = lo[i]; r[i + 8] = hi[i]; }
  return r;
}

__device__ __forceinline__ v8f wmma16(v16h a, v16h b, v8f c) {
  return __builtin_amdgcn_wmma_f32_16x16x32_f16(false, a, false, b, (short)0, c, false, false);
}

// ---------------------------------------------------------------------------
// Tiled transpose + f32->f16 convert: Wt[n][k] = (f16)W[k][n]. K,N multiples of 32.
// ---------------------------------------------------------------------------
__global__ __launch_bounds__(256) void transpose_f16_kernel(
    const float* __restrict__ W, _Float16* __restrict__ Wt, int K, int N) {
  __shared__ _Float16 t[32][33];
  const int n0 = blockIdx.x << 5, k0 = blockIdx.y << 5;
  const int tx = threadIdx.x & 31, ty = threadIdx.x >> 5;  // 32 x 8
#pragma unroll
  for (int i = 0; i < 32; i += 8)
    t[ty + i][tx] = (_Float16)W[(size_t)(k0 + ty + i) * N + n0 + tx];
  __syncthreads();
#pragma unroll
  for (int i = 0; i < 32; i += 8)
    Wt[(size_t)(n0 + ty + i) * K + k0 + tx] = t[tx][ty + i];
}

// ---------------------------------------------------------------------------
// Embedding gather
// ---------------------------------------------------------------------------
__global__ void embed_kernel(const int* __restrict__ ids, const float* __restrict__ embed,
                             float* __restrict__ x, int H) {
  int idx = blockIdx.x * blockDim.x + threadIdx.x;
  int t = idx / H, i = idx - t * H;
  x[idx] = embed[(size_t)ids[t] * H + i];
}

// ---------------------------------------------------------------------------
// RMSNorm: writes f16 and/or fp32
// ---------------------------------------------------------------------------
__global__ __launch_bounds__(256) void rmsnorm_kernel(const float* __restrict__ X,
                                                      const float* __restrict__ w,
                                                      _Float16* __restrict__ Y16,
                                                      float* __restrict__ Yf, int H) {
  const int rowi = blockIdx.x;
  const float* x = X + (size_t)rowi * H;
  __shared__ float red[8];
  float ss = 0.0f;
  for (int i = threadIdx.x; i < H; i += 256) { float v = x[i]; ss += v * v; }
#pragma unroll
  for (int o = 16; o > 0; o >>= 1) ss += __shfl_xor(ss, o, 32);
  if ((threadIdx.x & 31) == 0) red[threadIdx.x >> 5] = ss;
  __syncthreads();
  if (threadIdx.x == 0) {
    float t = 0.0f;
#pragma unroll
    for (int i = 0; i < 8; ++i) t += red[i];
    red[0] = rsqrtf(t / (float)H + 1e-6f);
  }
  __syncthreads();
  const float rs = red[0];
  for (int i = threadIdx.x; i < H; i += 256) {
    float v = x[i] * rs * w[i];
    if (Y16) Y16[(size_t)rowi * H + i] = (_Float16)v;
    if (Yf)  Yf[(size_t)rowi * H + i] = v;
  }
}

// ---------------------------------------------------------------------------
// Direct-global WMMA GEMM, ping-pong double-buffered (two fixed fragment sets,
// no rotation copies). acc = A16[M,K] @ Wt16[N,K]^T. No LDS, no barriers.
// 256 threads = 8 waves; wave w -> rows m0 = by*128 + w*16; 4 N-tiles per wave.
// Requires K/32 even (true for all uses: 1536, 8960).
// OUT: 0 = C32[m][n] = acc + res[m][n]; 1 = C16[m][n] = acc + bias;
//      2 = C16[n][m] = acc + bias (transposed, for V).
// ---------------------------------------------------------------------------
template <int OUT>
__global__ __launch_bounds__(256) void gemm_wmma_direct(
    const _Float16* __restrict__ A, const _Float16* __restrict__ Wt,
    const float* __restrict__ bias, const float* __restrict__ res,
    float* __restrict__ C, _Float16* __restrict__ C16,
    int M, int N, int K) {
  const int tid = threadIdx.x, wave = tid >> 5, lane = tid & 31;
  const int m0 = (blockIdx.y << 7) + (wave << 4);
  const int n0 = blockIdx.x << 6;
  const int row = lane & 15;
  const _Float16* arow = A + (size_t)(m0 + row) * K;
  const _Float16* b0 = Wt + (size_t)(n0 + row) * K;
  const _Float16* b1 = Wt + (size_t)(n0 + 16 + row) * K;
  const _Float16* b2 = Wt + (size_t)(n0 + 32 + row) * K;
  const _Float16* b3 = Wt + (size_t)(n0 + 48 + row) * K;
  v8f acc[4] = {};
  // set 0 / set 1 fragment buffers (ping-pong; never copied between sets)
  v16h a0, p00, p01, p02, p03;
  v16h a1, p10, p11, p12, p13;
  a0  = load_fragA(arow, lane);
  p00 = load_fragB(b0, lane);
  p01 = load_fragB(b1, lane);
  p02 = load_fragB(b2, lane);
  p03 = load_fragB(b3, lane);
  int k0 = 32;
  for (; k0 < K - 32; k0 += 64) {
    a1  = load_fragA(arow + k0, lane);
    p10 = load_fragB(b0 + k0, lane);
    p11 = load_fragB(b1 + k0, lane);
    p12 = load_fragB(b2 + k0, lane);
    p13 = load_fragB(b3 + k0, lane);
    acc[0] = wmma16(a0, p00, acc[0]);
    acc[1] = wmma16(a0, p01, acc[1]);
    acc[2] = wmma16(a0, p02, acc[2]);
    acc[3] = wmma16(a0, p03, acc[3]);
    a0  = load_fragA(arow + k0 + 32, lane);
    p00 = load_fragB(b0 + k0 + 32, lane);
    p01 = load_fragB(b1 + k0 + 32, lane);
    p02 = load_fragB(b2 + k0 + 32, lane);
    p03 = load_fragB(b3 + k0 + 32, lane);
    acc[0] = wmma16(a1, p10, acc[0]);
    acc[1] = wmma16(a1, p11, acc[1]);
    acc[2] = wmma16(a1, p12, acc[2]);
    acc[3] = wmma16(a1, p13, acc[3]);
  }
  // tail: one remaining step (K/32 even)
  a1  = load_fragA(arow + K - 32, lane);
  p10 = load_fragB(b0 + K - 32, lane);
  p11 = load_fragB(b1 + K - 32, lane);
  p12 = load_fragB(b2 + K - 32, lane);
  p13 = load_fragB(b3 + K - 32, lane);
  acc[0] = wmma16(a0, p00, acc[0]);
  acc[1] = wmma16(a0, p01, acc[1]);
  acc[2] = wmma16(a0, p02, acc[2]);
  acc[3] = wmma16(a0, p03, acc[3]);
  acc[0] = wmma16(a1, p10, acc[0]);
  acc[1] = wmma16(a1, p11, acc[1]);
  acc[2] = wmma16(a1, p12, acc[2]);
  acc[3] = wmma16(a1, p13, acc[3]);

  const int col = lane & 15;
  const int rbase = (lane >> 4) << 3;
#pragma unroll
  for (int t = 0; t < 4; ++t) {
    const int n = n0 + (t << 4) + col;
    const float bv = (OUT != 0) ? bias[n] : 0.0f;
#pragma unroll
    for (int r = 0; r < 8; ++r) {
      const int m = m0 + rbase + r;
      const float v = acc[t][r] + bv;
      if (OUT == 0)      C[(size_t)m * N + n] = v + res[(size_t)m * N + n];
      else if (OUT == 1) C16[(size_t)m * N + n] = (_Float16)v;
      else               C16[(size_t)n * M + m] = (_Float16)v;
    }
  }
}

// ---------------------------------------------------------------------------
// Fused gate+up GEMM (direct-global, ping-pong): act16 = silu(A@Wg) * (A@Wu)
// ---------------------------------------------------------------------------
__global__ __launch_bounds__(256) void gateup_direct(
    const _Float16* __restrict__ A, const _Float16* __restrict__ Wgt,
    const _Float16* __restrict__ Wut, _Float16* __restrict__ Act,
    int M, int N, int K) {
  const int tid = threadIdx.x, wave = tid >> 5, lane = tid & 31;
  const int m0 = (blockIdx.y << 7) + (wave << 4);
  const int n0 = blockIdx.x << 6;
  const int row = lane & 15;
  const _Float16* arow = A + (size_t)(m0 + row) * K;
  const _Float16* gb[4];
  const _Float16* ub[4];
#pragma unroll
  for (int t = 0; t < 4; ++t) {
    gb[t] = Wgt + (size_t)(n0 + (t << 4) + row) * K;
    ub[t] = Wut + (size_t)(n0 + (t << 4) + row) * K;
  }
  v8f ag[4] = {}, au[4] = {};
  v16h a0, g0[4], u0[4];
  v16h a1, g1[4], u1[4];
  a0 = load_fragA(arow, lane);
#pragma unroll
  for (int t = 0; t < 4; ++t) { g0[t] = load_fragB(gb[t], lane); u0[t] = load_fragB(ub[t], lane); }
  int k0 = 32;
  for (; k0 < K - 32; k0 += 64) {
    a1 = load_fragA(arow + k0, lane);
#pragma unroll
    for (int t = 0; t < 4; ++t) { g1[t] = load_fragB(gb[t] + k0, lane); u1[t] = load_fragB(ub[t] + k0, lane); }
#pragma unroll
    for (int t = 0; t < 4; ++t) {
      ag[t] = wmma16(a0, g0[t], ag[t]);
      au[t] = wmma16(a0, u0[t], au[t]);
    }
    a0 = load_fragA(arow + k0 + 32, lane);
#pragma unroll
    for (int t = 0; t < 4; ++t) { g0[t] = load_fragB(gb[t] + k0 + 32, lane); u0[t] = load_fragB(ub[t] + k0 + 32, lane); }
#pragma unroll
    for (int t = 0; t < 4; ++t) {
      ag[t] = wmma16(a1, g1[t], ag[t]);
      au[t] = wmma16(a1, u1[t], au[t]);
    }
  }
  a1 = load_fragA(arow + K - 32, lane);
#pragma unroll
  for (int t = 0; t < 4; ++t) { g1[t] = load_fragB(gb[t] + K - 32, lane); u1[t] = load_fragB(ub[t] + K - 32, lane); }
#pragma unroll
  for (int t = 0; t < 4; ++t) {
    ag[t] = wmma16(a0, g0[t], ag[t]);
    au[t] = wmma16(a0, u0[t], au[t]);
  }
#pragma unroll
  for (int t = 0; t < 4; ++t) {
    ag[t] = wmma16(a1, g1[t], ag[t]);
    au[t] = wmma16(a1, u1[t], au[t]);
  }
  const int col = lane & 15;
  const int rbase = (lane >> 4) << 3;
#pragma unroll
  for (int t = 0; t < 4; ++t) {
    const int n = n0 + (t << 4) + col;
#pragma unroll
    for (int r = 0; r < 8; ++r) {
      const int m = m0 + rbase + r;
      const float g = ag[t][r], u = au[t][r];
      Act[(size_t)m * N + n] = (_Float16)(g / (1.0f + __expf(-g)) * u);
    }
  }
}

// ---------------------------------------------------------------------------
// RoPE in-place on f16 [rows = B*S*NHx, 128]
// ---------------------------------------------------------------------------
__global__ void rope_f16_kernel(_Float16* __restrict__ X16, int S, int NHx, float theta) {
  const int DH = 128;
  int idx = blockIdx.x * blockDim.x + threadIdx.x;
  int d = idx & 63;
  int rest = idx >> 6;                       // ((b*S + t)*NHx + h)
  int tpos = (rest / NHx) % S;
  float inv = __expf(-((float)(2 * d) / (float)DH) * __logf(theta));
  float ang = (float)tpos * inv;
  float c = __cosf(ang), s = __sinf(ang);
  _Float16* p = X16 + (size_t)rest * DH;
  float x0 = (float)p[d], x1 = (float)p[d + 64];
  p[d]      = (_Float16)(x0 * c - x1 * s);
  p[d + 64] = (_Float16)(x1 * c + x0 * s);
}

// ---------------------------------------------------------------------------
// Flash attention (causal, GQA). Wave per 16 query rows; WMMA QK^T and PV.
// K fragments batch-loaded per key block; V fragments issued before the
// softmax barrier so they overlap the softmax latency.
// ---------------------------------------------------------------------------
__global__ __launch_bounds__(128) void attn_kernel(
    const _Float16* __restrict__ Q,  // [B,S,NH,DH]
    const _Float16* __restrict__ Kc, // [B,S,NKV,DH]
    const _Float16* __restrict__ Vt, // [NKV*DH, B*S]
    _Float16* __restrict__ O,        // [B,S,NH*DH]
    int S, int BS) {
  const int NH = 12, NKV = 2, DH = 128;
  const float scale = 0.088388347648318447f;  // 1/sqrt(128)
  __shared__ float    sS[4][16 * 32];
  __shared__ float    sRow[4][16];
  __shared__ _Float16 sP[4][16 * 32];
  const int tid = threadIdx.x, wave = tid >> 5, lane = tid & 31;
  int gid = blockIdx.x;
  const int qBlocks = S >> 6;
  const int qgrp = gid % qBlocks; gid /= qBlocks;
  const int h = gid % NH;
  const int b = gid / NH;
  const int hk = h / (NH / NKV);
  const int m0 = (qgrp << 6) + (wave << 4);
  const int row = lane & 15, col = lane & 15;
  const int rbase = (lane >> 4) << 3;

  const _Float16* qrow = Q + ((size_t)(b * S + m0 + row) * NH + h) * DH;
  v16h qf[4];
#pragma unroll
  for (int kc = 0; kc < 4; ++kc) qf[kc] = load_fragA(qrow + kc * 32, lane);

  v8f oacc[8] = {};
  float m_run = -1e30f, l_run = 0.0f;
  const int kend = (qgrp << 6) + 64;          // uniform across the 4 waves
  for (int kb = 0; kb < kend; kb += 32) {
    // --- batch-load all K fragments for this key block (2 tiles x 4 chunks)
    v16h kf[8];
#pragma unroll
    for (int t = 0; t < 2; ++t) {
      const int key = kb + (t << 4) + col;
      const _Float16* krow = Kc + ((size_t)(b * S + key) * NKV + hk) * DH;
#pragma unroll
      for (int kc = 0; kc < 4; ++kc) kf[t * 4 + kc] = load_fragB(krow + kc * 32, lane);
    }
    // --- V fragments don't depend on softmax: issue now to hide latency
    v16h vfs[8];
#pragma unroll
    for (int t = 0; t < 8; ++t) {
      const int dh = (t << 4) + col;
      vfs[t] = load_fragB(Vt + (size_t)(hk * DH + dh) * BS + b * S + kb, lane);
    }
    // --- scores
#pragma unroll
    for (int t = 0; t < 2; ++t) {
      v8f s = {};
#pragma unroll
      for (int kc = 0; kc < 4; ++kc) s = wmma16(qf[kc], kf[t * 4 + kc], s);
#pragma unroll
      for (int r = 0; r < 8; ++r)
        sS[wave][(rbase + r) * 32 + (t << 4) + col] = s[r] * scale;
    }
    __syncthreads();
    // --- per-row online softmax (lanes 0..15)
    if (lane < 16) {
      const int qi = m0 + row;
      float mnew = m_run;
      float sc[32];
#pragma unroll
      for (int c = 0; c < 32; ++c) {
        float v = sS[wave][row * 32 + c];
        if (kb + c > qi) v = -1e30f;
        sc[c] = v;
        mnew = fmaxf(mnew, v);
      }
      const float alpha = __expf(m_run - mnew);
      float ls = 0.0f;
#pragma unroll
      for (int c = 0; c < 32; ++c) {
        float p = __expf(sc[c] - mnew);
        ls += p;
        sP[wave][row * 32 + c] = (_Float16)p;
      }
      l_run = l_run * alpha + ls;
      m_run = mnew;
      sRow[wave][row] = alpha;
    }
    __syncthreads();
    // --- rescale O by alpha(row)
#pragma unroll
    for (int r = 0; r < 8; ++r) {
      const float a = sRow[wave][rbase + r];
#pragma unroll
      for (int t = 0; t < 8; ++t) oacc[t][r] *= a;
    }
    // --- O += P @ V (V fragments already resident)
    v16h pf = load_fragA(&sP[wave][row * 32], lane);
#pragma unroll
    for (int t = 0; t < 8; ++t) oacc[t] = wmma16(pf, vfs[t], oacc[t]);
    __syncthreads();
  }
  if (lane < 16) sRow[wave][row] = l_run;
  __syncthreads();
#pragma unroll
  for (int r = 0; r < 8; ++r) {
    const float invl = 1.0f / sRow[wave][rbase + r];
#pragma unroll
    for (int t = 0; t < 8; ++t)
      O[((size_t)(b * S + m0 + rbase + r)) * (NH * DH) + h * DH + (t << 4) + col] =
          (_Float16)(oacc[t][r] * invl);
  }
}

// ---------------------------------------------------------------------------
// Last-token gather
// ---------------------------------------------------------------------------
__global__ void last_hidden_kernel(const int* __restrict__ mask,
                                   const float* __restrict__ hidden,
                                   float* __restrict__ out, int S, int H) {
  const int b = blockIdx.x;
  __shared__ int sl;
  if (threadIdx.x == 0) {
    int c = 0;
    for (int s = 0; s < S; ++s) c += (mask[b * S + s] > 0) ? 1 : 0;
    sl = c - 1;
  }
  __syncthreads();
  for (int i = threadIdx.x; i < H; i += blockDim.x)
    out[(size_t)b * H + i] = hidden[((size_t)b * S + sl) * H + i];
}

// ---------------------------------------------------------------------------
extern "C" void kernel_launch(void* const* d_in, const int* in_sizes, int n_in,
                              void* d_out, int out_size, void* d_ws, size_t ws_size,
                              hipStream_t stream) {
  (void)in_sizes; (void)n_in; (void)out_size; (void)ws_size;
  const int* ids   = (const int*)d_in[0];
  const int* mask  = (const int*)d_in[1];
  const float* emb = (const float*)d_in[2];
  const float* wq  = (const float*)d_in[3];
  const float* bq  = (const float*)d_in[4];
  const float* wk  = (const float*)d_in[5];
  const float* bk  = (const float*)d_in[6];
  const float* wv  = (const float*)d_in[7];
  const float* bv  = (const float*)d_in[8];
  const float* wo  = (const float*)d_in[9];
  const float* wg  = (const float*)d_in[10];
  const float* wu  = (const float*)d_in[11];
  const float* wd  = (const float*)d_in[12];
  const float* ln1 = (const float*)d_in[13];
  const float* ln2 = (const float*)d_in[14];
  const float* lnf = (const float*)d_in[15];
  float* out = (float*)d_out;

  const int B = 2, S = 1024, L = 2, NH = 12, NKV = 2, DH = 128, H = 1536, F = 8960;
  const int qd = NH * DH, kvd = NKV * DH, M = B * S;

  char* ws = (char*)d_ws;
  size_t off = 0;
  auto alloc = [&](size_t bytes) {
    void* p = ws + off;
    off = (off + bytes + 255) & ~(size_t)255;
    return p;
  };
  float*    x     = (float*)alloc((size_t)M * H * 4);
  _Float16* h16   = (_Float16*)alloc((size_t)M * H * 2);
  _Float16* q16   = (_Float16*)alloc((size_t)M * qd * 2);
  _Float16* k16   = (_Float16*)alloc((size_t)M * kvd * 2);
  _Float16* vT    = (_Float16*)alloc((size_t)M * kvd * 2);
  _Float16* o16   = (_Float16*)alloc((size_t)M * qd * 2);
  _Float16* act16 = (_Float16*)alloc((size_t)M * F * 2);
  _Float16* Wt1   = (_Float16*)alloc((size_t)H * F * 2);   // largest weight, reused
  _Float16* Wt2   = (_Float16*)alloc((size_t)H * F * 2);

  auto tr = [&](const float* W, int K, int N, _Float16* dst) {
    transpose_f16_kernel<<<dim3(N / 32, K / 32), 256, 0, stream>>>(W, dst, K, N);
  };

  embed_kernel<<<dim3((M * H) / 256), 256, 0, stream>>>(ids, emb, x, H);

  for (int l = 0; l < L; ++l) {
    rmsnorm_kernel<<<M, 256, 0, stream>>>(x, ln1 + l * H, h16, (float*)nullptr, H);
    tr(wq + (size_t)l * H * qd, H, qd, Wt1);
    gemm_wmma_direct<1><<<dim3(qd / 64, M / 128), 256, 0, stream>>>(
        h16, Wt1, bq + l * qd, nullptr, nullptr, q16, M, qd, H);
    tr(wk + (size_t)l * H * kvd, H, kvd, Wt1);
    gemm_wmma_direct<1><<<dim3(kvd / 64, M / 128), 256, 0, stream>>>(
        h16, Wt1, bk + l * kvd, nullptr, nullptr, k16, M, kvd, H);
    tr(wv + (size_t)l * H * kvd, H, kvd, Wt1);
    gemm_wmma_direct<2><<<dim3(kvd / 64, M / 128), 256, 0, stream>>>(
        h16, Wt1, bv + l * kvd, nullptr, nullptr, vT, M, kvd, H);
    rope_f16_kernel<<<dim3((M * NH * 64) / 256), 256, 0, stream>>>(q16, S, NH, 1000000.0f);
    rope_f16_kernel<<<dim3((M * NKV * 64) / 256), 256, 0, stream>>>(k16, S, NKV, 1000000.0f);
    attn_kernel<<<dim3(B * NH * (S / 64)), 128, 0, stream>>>(q16, k16, vT, o16, S, M);
    tr(wo + (size_t)l * qd * H, qd, H, Wt1);
    gemm_wmma_direct<0><<<dim3(H / 64, M / 128), 256, 0, stream>>>(
        o16, Wt1, nullptr, x, x, nullptr, M, H, qd);
    rmsnorm_kernel<<<M, 256, 0, stream>>>(x, ln2 + l * H, h16, (float*)nullptr, H);
    tr(wg + (size_t)l * H * F, H, F, Wt1);
    tr(wu + (size_t)l * H * F, H, F, Wt2);
    gateup_direct<<<dim3(F / 64, M / 128), 256, 0, stream>>>(h16, Wt1, Wt2, act16, M, F, H);
    tr(wd + (size_t)l * F * H, F, H, Wt1);
    gemm_wmma_direct<0><<<dim3(H / 64, M / 128), 256, 0, stream>>>(
        act16, Wt1, nullptr, x, x, nullptr, M, H, F);
  }

  rmsnorm_kernel<<<M, 256, 0, stream>>>(x, lnf, (_Float16*)nullptr, out, H);
  last_hidden_kernel<<<B, 256, 0, stream>>>(mask, out, out + (size_t)M * H, S, H);
}
